// SaltMHA_27857157882495
// MI455X (gfx1250) — matrix operand order repaired and checked by reading, not compiled, not run on hardware
//
#include <hip/hip_runtime.h>

// ---------------------------------------------------------------------------
// Problem constants (from reference): B=4, S=1024, E=512, H=8, D=64
// ---------------------------------------------------------------------------
constexpr int Bc = 4, Sc = 1024, Ec = 512, Hc = 8, Dc = 64;
constexpr int BH = Bc * Hc;   // 32

typedef __attribute__((ext_vector_type(16))) __bf16        v16bf;
typedef __attribute__((ext_vector_type(8)))  __bf16        v8bf;
typedef __attribute__((ext_vector_type(8)))  float         v8f;
typedef __attribute__((ext_vector_type(4)))  unsigned int  v4u;

// float -> bf16 (round to nearest even), stored as u16
__device__ inline unsigned short f2bf(float f) {
    unsigned int u = __builtin_bit_cast(unsigned int, f);
    unsigned int r = u + 0x7FFFu + ((u >> 16) & 1u);
    return (unsigned short)(r >> 16);
}

__device__ inline v16bf cat16(v8bf a, v8bf b) {
    return __builtin_shufflevector(a, b, 0,1,2,3,4,5,6,7,8,9,10,11,12,13,14,15);
}

// A fragment (16x32 bf16, M=row, K=d): lane<16 -> K 0..15,
// lane>=16 -> K 16..31.  base = rowptr + dbase + half*16
__device__ inline v16bf load_a_frag(const unsigned short* base) {
    v8bf c0 = *reinterpret_cast<const v8bf*>(base);
    v8bf c1 = *reinterpret_cast<const v8bf*>(base + 8);
    return cat16(c0, c1);
}
// B fragment (32x16 bf16, K, N=col): lane<16 -> K 0..15, lane>=16 -> K 16..31.
// base = ptr to the N-th K-major row + kbase + half*16
__device__ inline v16bf load_b_frag(const unsigned short* base) {
    v8bf c0 = *reinterpret_cast<const v8bf*>(base);
    v8bf c1 = *reinterpret_cast<const v8bf*>(base + 8);
    return cat16(c0, c1);
}

__device__ inline v8f wmma_bf16(v16bf a, v16bf b, v8f c) {
    return __builtin_amdgcn_wmma_f32_16x16x32_bf16(
        /*neg_a=*/false, a, /*neg_b=*/false, b,
        /*c_mod=*/(short)0, c, /*reuse_a=*/false, /*reuse_b=*/false);
}

// ---------------------------------------------------------------------------
// Kernel 1: split heads + convert fp32 -> bf16:  QB/KB: (BH, S, D) bf16
// ---------------------------------------------------------------------------
__global__ void __launch_bounds__(256)
prep_kernel(const float* __restrict__ src, const float* __restrict__ tgt,
            unsigned short* __restrict__ QB, unsigned short* __restrict__ KB) {
    int i = blockIdx.x * 256 + threadIdx.x;           // < B*S*E
    int e = i % Ec;
    int s = (i / Ec) % Sc;
    int b = i / (Ec * Sc);
    int h = e / Dc, d = e % Dc;
    size_t qk = (((size_t)(b * Hc + h)) * Sc + s) * Dc + d;
    QB[qk] = f2bf(src[i]);
    KB[qk] = f2bf(tgt[i]);
}

// ---------------------------------------------------------------------------
// Kernel 2: one-shot transpose target (B,S,E) f32 -> TT (B,E,S) bf16.
// 64x64 tiles staged into LDS with ASYNC global->LDS copies (ASYNCcnt path,
// no VGPR round-trip), then column reads from padded-pitch LDS, bf16-pack,
// coalesced 32B row writes.  TT is 4MB total -> lives in L2 for kernel 5.
// ---------------------------------------------------------------------------
__global__ void __launch_bounds__(256)
transpose_kernel(const float* __restrict__ tgt, unsigned short* __restrict__ TT) {
    constexpr int PITCH = 68;                       // floats; breaks bank conflicts
    __shared__ __align__(16) float lds_f[64 * PITCH];
    int b  = blockIdx.z;
    int s0 = blockIdx.y * 64;
    int e0 = blockIdx.x * 64;
    int tid = threadIdx.x;

    // phase 1: async-copy 64 rows x 64 f32 (16 b128 chunks per row) into LDS
    unsigned ldsBase = (unsigned)(size_t)(&lds_f[0]);
#pragma unroll
    for (int i = 0; i < 4; ++i) {
        int c    = tid + i * 256;                   // 0..1023 chunk id
        int srow = c >> 4;                          // 0..63
        int cb   = (c & 15) * 16;                   // byte offset within row
        const float* gp = tgt + ((size_t)(b * Sc + s0 + srow)) * Ec + e0;
        unsigned long long ga = (unsigned long long)gp + (unsigned)cb;
        unsigned la = ldsBase + (unsigned)(srow * PITCH * 4 + cb);
        asm volatile("global_load_async_to_lds_b128 %0, %1, off"
                     :: "v"(la), "v"(ga) : "memory");
    }
    asm volatile("s_wait_asynccnt 0x0" ::: "memory");
    __syncthreads();

    // phase 2: each thread emits one 16-wide bf16 run of a transposed row
    int e_l  = tid >> 2;                            // 0..63
    int sBlk = (tid & 3) * 16;                      // 0..48
    unsigned int w[8];
#pragma unroll
    for (int k = 0; k < 8; ++k) {
        unsigned lo = f2bf(lds_f[(sBlk + 2 * k)     * PITCH + e_l]);
        unsigned hi = f2bf(lds_f[(sBlk + 2 * k + 1) * PITCH + e_l]);
        w[k] = lo | (hi << 16);
    }
    unsigned short* dst = TT + ((size_t)(b * Ec + e0 + e_l)) * Sc + s0 + sBlk;
    *reinterpret_cast<v4u*>(dst)     = (v4u){w[0], w[1], w[2], w[3]};
    *reinterpret_cast<v4u*>(dst + 8) = (v4u){w[4], w[5], w[6], w[7]};
}

// ---------------------------------------------------------------------------
// Kernel 3: the sequential scan.  2048 lanes (b,h,d), 1024 tanh steps each.
// ---------------------------------------------------------------------------
__global__ void __launch_bounds__(256)
recur_kernel(const float* __restrict__ tgt, const float* __restrict__ enc,
             unsigned short* __restrict__ CCB) {
    int tid = blockIdx.x * 256 + threadIdx.x;
    if (tid >= BH * Dc) return;
    int d = tid % Dc;
    int h = (tid / Dc) % Hc;
    int b = tid / (Dc * Hc);
    float pe = enc[h * Dc + d];
    const float* tb = tgt + (size_t)b * Sc * Ec + h * Dc + d;
    unsigned short* cc = CCB + ((size_t)(b * Hc + h)) * Sc * Dc + d;
    float c = tb[0];
    for (int t = 0; t < Sc; ++t) {
        float v = tanhf(c * pe);
        cc[(size_t)t * Dc] = f2bf(v);
        c = tb[(size_t)t * Ec] * v;
    }
}

// ---------------------------------------------------------------------------
// Kernel 4: per-row softmax denominators.  One wave per 16 q-rows.
//   Sub-pass 1: Zg[s] = sum_t exp((q.cc_t)/8)
//   Sub-pass 2: Zw[s] = sum_t exp((a*(q.k_t)/100 + b*exp(gl)/Zg)/g)
// Logits are O(1) so exp without max-subtraction is safe.
// ---------------------------------------------------------------------------
__global__ void __launch_bounds__(32)
rowsum_kernel(const unsigned short* __restrict__ QB,
              const unsigned short* __restrict__ KB,
              const unsigned short* __restrict__ CCB,
              const float* __restrict__ pA, const float* __restrict__ pB,
              const float* __restrict__ pG,
              float* __restrict__ Zginv, float* __restrict__ Zwinv) {
    int bh = blockIdx.y;
    int s0 = blockIdx.x * 16;
    int lane = threadIdx.x & 31, half = lane >> 4, m = lane & 15;
    float alpha = pA[0], beta = pB[0], invG = 1.0f / pG[0];

    const unsigned short* qrow = QB + ((size_t)(bh * Sc + s0 + m)) * Dc;
    v16bf a0 = load_a_frag(qrow + 0  + half * 16);
    v16bf a1 = load_a_frag(qrow + 32 + half * 16);

    float zg[8];
#pragma unroll
    for (int j = 0; j < 8; ++j) zg[j] = 0.0f;

    for (int tt = 0; tt < Sc / 16; ++tt) {
        const unsigned short* crow = CCB + ((size_t)(bh * Sc + tt * 16 + m)) * Dc;
        v16bf b0 = load_b_frag(crow + 0  + half * 16);
        v16bf b1 = load_b_frag(crow + 32 + half * 16);
        v8f acc = {};
        acc = wmma_bf16(a0, b0, acc);
        acc = wmma_bf16(a1, b1, acc);
#pragma unroll
        for (int j = 0; j < 8; ++j) zg[j] += __expf(acc[j] * 0.125f);
    }
#pragma unroll
    for (int j = 0; j < 8; ++j) {
        for (int msk = 1; msk <= 8; msk <<= 1) zg[j] += __shfl_xor(zg[j], msk, 32);
        zg[j] = 1.0f / zg[j];       // Zg^-1, identical across the 16-lane half
    }

    float zw[8];
#pragma unroll
    for (int j = 0; j < 8; ++j) zw[j] = 0.0f;

    for (int tt = 0; tt < Sc / 16; ++tt) {
        const unsigned short* crow = CCB + ((size_t)(bh * Sc + tt * 16 + m)) * Dc;
        const unsigned short* krow = KB  + ((size_t)(bh * Sc + tt * 16 + m)) * Dc;
        v16bf bc0 = load_b_frag(crow + 0  + half * 16);
        v16bf bc1 = load_b_frag(crow + 32 + half * 16);
        v16bf bk0 = load_b_frag(krow + 0  + half * 16);
        v16bf bk1 = load_b_frag(krow + 32 + half * 16);
        v8f g = {}, sm = {};
        g  = wmma_bf16(a0, bc0, g);  g  = wmma_bf16(a1, bc1, g);
        sm = wmma_bf16(a0, bk0, sm); sm = wmma_bf16(a1, bk1, sm);
#pragma unroll
        for (int j = 0; j < 8; ++j) {
            float G  = __expf(g[j] * 0.125f) * zg[j];
            float wl = (alpha * sm[j] * 0.01f + beta * G) * invG;
            zw[j] += __expf(wl);
        }
    }
#pragma unroll
    for (int j = 0; j < 8; ++j)
        for (int msk = 1; msk <= 8; msk <<= 1) zw[j] += __shfl_xor(zw[j], msk, 32);

    if (m == 0) {
#pragma unroll
        for (int j = 0; j < 8; ++j) {
            int s = s0 + j + half * 8;
            Zginv[bh * Sc + s] = zg[j];
            Zwinv[bh * Sc + s] = 1.0f / zw[j];
        }
    }
}

// ---------------------------------------------------------------------------
// Kernel 5: out[bh, t, e] = sum_s W[s,t] * target[s,e]
// Workgroup = 4 waves, 64 t-rows x 128 e-cols of one (b,h); no barriers.
// Per 32-wide s-chunk each wave recomputes its (t x s) logit tile TRANSPOSED
// (M=t, N=s) so Zg^-1[s]/Zw^-1[s] are per-lane scalars, applies both
// softmaxes, round-trips W through a per-wave LDS tile into an A fragment,
// and runs one accumulation WMMA per e-tile with B fragments loaded straight
// from the pre-transposed TT (L2-resident) as contiguous global_load_b128s.
// ---------------------------------------------------------------------------
__global__ void __launch_bounds__(128)
out_kernel(const unsigned short* __restrict__ QB,
           const unsigned short* __restrict__ KB,
           const unsigned short* __restrict__ CCB,
           const unsigned short* __restrict__ TT,
           const float* __restrict__ Zginv, const float* __restrict__ Zwinv,
           const float* __restrict__ pA, const float* __restrict__ pB,
           const float* __restrict__ pG, float* __restrict__ out) {
    __shared__ __align__(16) unsigned short lds_w[4][16][32]; // per-wave (t,s) W

    int bh   = blockIdx.z;
    int wave = threadIdx.x >> 5, lane = threadIdx.x & 31;
    int half = lane >> 4, m = lane & 15;
    int t0 = blockIdx.y * 64 + wave * 16;
    int e0 = blockIdx.x * 128;
    float alpha = pA[0], beta = pB[0], invG = 1.0f / pG[0];
    int b = bh / Hc;

    // A fragments of K and CC rows (M = t) are loop-invariant
    const unsigned short* krow = KB  + ((size_t)(bh * Sc + t0 + m)) * Dc;
    const unsigned short* crow = CCB + ((size_t)(bh * Sc + t0 + m)) * Dc;
    v16bf ak0 = load_a_frag(krow + 0  + half * 16);
    v16bf ak1 = load_a_frag(krow + 32 + half * 16);
    v16bf ac0 = load_a_frag(crow + 0  + half * 16);
    v16bf ac1 = load_a_frag(crow + 32 + half * 16);

    v8f acc[8];
#pragma unroll
    for (int et = 0; et < 8; ++et) acc[et] = (v8f){};

    for (int sc = 0; sc < Sc; sc += 32) {
        // W tile (16 t x 32 s) from two transposed 16x16 logit tiles
#pragma unroll
        for (int sub = 0; sub < 2; ++sub) {
            int sBase = sc + sub * 16;
            const unsigned short* qrow = QB + ((size_t)(bh * Sc + sBase + m)) * Dc;
            v16bf bq0 = load_b_frag(qrow + 0  + half * 16);
            v16bf bq1 = load_b_frag(qrow + 32 + half * 16);
            v8f g = {}, sm = {};
            g  = wmma_bf16(ac0, bq0, g);  g  = wmma_bf16(ac1, bq1, g);
            sm = wmma_bf16(ak0, bq0, sm); sm = wmma_bf16(ak1, bq1, sm);
            int sIdx  = sBase + m;                 // N = s: per-lane scalar
            float zgi = Zginv[bh * Sc + sIdx];
            float zwi = Zwinv[bh * Sc + sIdx];
#pragma unroll
            for (int j = 0; j < 8; ++j) {
                float G  = __expf(g[j] * 0.125f) * zgi;
                float wl = (alpha * sm[j] * 0.01f + beta * G) * invG;
                float w  = __expf(wl) * zwi;
                lds_w[wave][j + half * 8][sub * 16 + m] = f2bf(w);
            }
        }
        asm volatile("s_wait_dscnt 0" ::: "memory");  // wave-local LDS RAW fence

        // A fragment of W^T (M=t, K=s 0..31) from the per-wave LDS tile
        const unsigned short* wr = &lds_w[wave][m][0];
        v8bf w0 = *reinterpret_cast<const v8bf*>(wr + half * 16);
        v8bf w1 = *reinterpret_cast<const v8bf*>(wr + half * 16 + 8);
        v16bf aw = cat16(w0, w1);

        // accumulate the 8 e-tiles; B fragments direct from L2-resident TT
#pragma unroll
        for (int et = 0; et < 8; ++et) {
            const unsigned short* tr =
                TT + ((size_t)(b * Ec + e0 + et * 16 + m)) * Sc + sc + half * 16;
            v8bf b0 = *reinterpret_cast<const v8bf*>(tr);
            v8bf b1 = *reinterpret_cast<const v8bf*>(tr + 8);
            acc[et] = wmma_bf16(aw, cat16(b0, b1), acc[et]);
        }
    }

    // write out[b, h*S + t, e]  ==  out[(bh*S + t)*E + e]
#pragma unroll
    for (int et = 0; et < 8; ++et)
#pragma unroll
        for (int j = 0; j < 8; ++j) {
            int t = t0 + j + half * 8;
            int e = e0 + et * 16 + m;
            out[((size_t)(bh * Sc + t)) * Ec + e] = acc[et][j];
        }
}

// ---------------------------------------------------------------------------
extern "C" void kernel_launch(void* const* d_in, const int* in_sizes, int n_in,
                              void* d_out, int out_size, void* d_ws, size_t ws_size,
                              hipStream_t stream) {
    const float* src = (const float*)d_in[0];   // source (B,S,E)
    const float* tgt = (const float*)d_in[1];   // target (B,S,E)
    const float* enc = (const float*)d_in[2];   // encoder (1,H,1,D)
    const float* pA  = (const float*)d_in[3];   // alpha
    const float* pB  = (const float*)d_in[4];   // beta
    const float* pG  = (const float*)d_in[5];   // gamma
    float* out = (float*)d_out;                 // (B, H*S, E)

    // workspace: QB/KB/CCB (BH,S,D) bf16 + TT (B,E,S) bf16 + Z's  (~16.25 MB)
    unsigned short* QB  = (unsigned short*)d_ws;
    unsigned short* KB  = QB  + (size_t)BH * Sc * Dc;
    unsigned short* CCB = KB  + (size_t)BH * Sc * Dc;
    unsigned short* TT  = CCB + (size_t)BH * Sc * Dc;
    float* Zginv = (float*)(TT + (size_t)Bc * Ec * Sc);
    float* Zwinv = Zginv + (size_t)BH * Sc;

    int nElem = Bc * Sc * Ec;                                   // 2,097,152
    prep_kernel<<<nElem / 256, 256, 0, stream>>>(src, tgt, QB, KB);
    transpose_kernel<<<dim3(Ec / 64, Sc / 64, Bc), 256, 0, stream>>>(tgt, TT);
    recur_kernel<<<(BH * Dc + 255) / 256, 256, 0, stream>>>(tgt, enc, CCB);
    rowsum_kernel<<<dim3(Sc / 16, BH), 32, 0, stream>>>(QB, KB, CCB,
                                                        pA, pB, pG, Zginv, Zwinv);
    out_kernel<<<dim3(Ec / 128, Sc / 64, BH), 128, 0, stream>>>(
        QB, KB, CCB, TT, Zginv, Zwinv, pA, pB, pG, out);
}
